// SpanAsteModel_83476984365370
// MI455X (gfx1250) — matrix operand order repaired and checked by reading, not compiled, not run on hardware
//
#include <hip/hip_runtime.h>
#include <hip/hip_bf16.h>

// ---------------------------------------------------------------------------
// Types for CDNA5 WMMA (wave32, 16x16x32 bf16 -> f32 accumulate)
// ---------------------------------------------------------------------------
typedef __attribute__((ext_vector_type(16))) __bf16         v16bf;
typedef __attribute__((ext_vector_type(2)))  __bf16         v2bf;
typedef __attribute__((ext_vector_type(2)))  float          v2f;
typedef __attribute__((ext_vector_type(8)))  float          v8f;
typedef __attribute__((ext_vector_type(8)))  unsigned short u16x8;

union FragU { v16bf bf; u16x8 h[2]; };

// f32 pair -> packed bf16 dword via vector fptrunc (RNE); the backend should
// select v_cvt_pk_bf16_f32 on gfx1250.
__device__ __forceinline__ unsigned int pk2bf(float lo, float hi) {
  v2f s; s.x = lo; s.y = hi;
  v2bf v = __builtin_convertvector(s, v2bf);
  unsigned int u;
  __builtin_memcpy(&u, &v, sizeof(u));
  return u;
}

__device__ __forceinline__ unsigned short f2bf(float f) {
  return (unsigned short)pk2bf(f, 0.f);
}

__device__ __forceinline__ v8f wmma_bf16(const FragU& a, const FragU& b, v8f c) {
  return __builtin_amdgcn_wmma_f32_16x16x32_bf16(
      /*neg_a=*/false, a.bf, /*neg_b=*/false, b.bf,
      /*c_mod=*/(short)0, c, /*reuse_a=*/false, /*reuse_b=*/false);
}

// span index -> (start, end); Ns = 630 spans over L=128, widths 1..5
__device__ __forceinline__ void span_se(int s, int* st, int* en) {
  int w = 1, rem = s;
  while (rem >= 129 - w) { rem -= 129 - w; ++w; }
  *st = rem; *en = rem + w - 1;
}

// ---------------------------------------------------------------------------
// Generic WMMA GEMM: C[M,N] = act(A[M,K] * W[N,K]^T + bias[N])
// A,W,C are f32 (lda/ldw/ldc all even); bf16 conversion while staging.
// WG = 256 threads (8 waves), tile 64(M) x 128(N), BK = 32, double-buffered.
// LDS tiles store 2 bf16 per dword: As[m][k/2], Bs[n][k/2].
// ---------------------------------------------------------------------------
#define GBM 64
#define GBN 128
#define GBK 32

__global__ __launch_bounds__(256) void k_gemm(
    const float* __restrict__ A, const float* __restrict__ W,
    const float* __restrict__ bias, float* __restrict__ C,
    int M, int N, int K, int lda, int ldw, int ldc, int relu) {
  __shared__ unsigned int As[2][GBM * GBK / 2];  // [m][k/2], row stride 16 dw
  __shared__ unsigned int Bs[2][GBK * GBN / 2];  // [n][k/2], row stride 16 dw

  const int tid  = threadIdx.x;
  const int wave = tid >> 5, lane = tid & 31;
  const int wm = wave & 1;     // 0..1  -> 32-row group
  const int wn = wave >> 1;    // 0..3  -> 32-col group
  const int m0 = blockIdx.y * GBM;
  const int n0 = blockIdx.x * GBN;

  // staging maps
  const int row_a = tid >> 2;            // 0..63
  const int ka    = (tid & 3) * 8;       // 0,8,16,24
  const int np    = tid & 63;            // n-pair 0..63 (n = 2*np)
  const int kg    = (tid >> 6) * 8;      // 0,8,16,24
  // fragment maps
  const int mrow = lane & 15, hi = lane >> 4;

  auto stageA = [&](int buf, int k0) {
    const int gm = m0 + row_a;
    const int kk = k0 + ka;
    uint4 w4;
    if ((m0 + GBM <= M) && (k0 + GBK <= K)) {
      const float* p = A + (size_t)gm * lda + kk;
      const float2 a0 = *(const float2*)(p + 0);
      const float2 a1 = *(const float2*)(p + 2);
      const float2 a2 = *(const float2*)(p + 4);
      const float2 a3 = *(const float2*)(p + 6);
      w4.x = pk2bf(a0.x, a0.y); w4.y = pk2bf(a1.x, a1.y);
      w4.z = pk2bf(a2.x, a2.y); w4.w = pk2bf(a3.x, a3.y);
    } else {
      const int gmc = gm < M ? gm : M - 1;
      const float* p = A + (size_t)gmc * lda;
      unsigned int t[4];
      #pragma unroll
      for (int j = 0; j < 4; ++j) {
        const int k1 = kk + 2 * j, k2 = k1 + 1;
        float v0 = p[k1 < K ? k1 : K - 1];
        float v1 = p[k2 < K ? k2 : K - 1];
        v0 = (gm < M && k1 < K) ? v0 : 0.f;
        v1 = (gm < M && k2 < K) ? v1 : 0.f;
        t[j] = pk2bf(v0, v1);
      }
      w4.x = t[0]; w4.y = t[1]; w4.z = t[2]; w4.w = t[3];
    }
    *(uint4*)&As[buf][row_a * 16 + (ka >> 1)] = w4;
  };

  auto stageB = [&](int buf, int k0) {
    const int gn0 = n0 + 2 * np, gn1 = gn0 + 1;
    if ((n0 + GBN <= N) && (k0 + GBK <= K)) {
      const float* p0 = W + (size_t)gn0 * ldw + k0 + kg;
      const float* p1 = W + (size_t)gn1 * ldw + k0 + kg;
      #pragma unroll
      for (int j = 0; j < 8; j += 2) {
        const float2 x0 = *(const float2*)(p0 + j);
        const float2 x1 = *(const float2*)(p1 + j);
        Bs[buf][(2 * np) * 16 + ((kg + j) >> 1)]     = pk2bf(x0.x, x0.y);
        Bs[buf][(2 * np + 1) * 16 + ((kg + j) >> 1)] = pk2bf(x1.x, x1.y);
      }
    } else {
      const int g0c = gn0 < N ? gn0 : N - 1;
      const int g1c = gn1 < N ? gn1 : N - 1;
      const float* p0 = W + (size_t)g0c * ldw;
      const float* p1 = W + (size_t)g1c * ldw;
      #pragma unroll
      for (int j = 0; j < 8; ++j) {
        const int kk = k0 + kg + j;
        const int kkc = kk < K ? kk : K - 1;
        float v0 = p0[kkc];
        float v1 = p1[kkc];
        v0 = (gn0 < N && kk < K) ? v0 : 0.f;
        v1 = (gn1 < N && kk < K) ? v1 : 0.f;
        ((unsigned short*)Bs[buf])[(2 * np) * 32 + kg + j] = f2bf(v0);
        ((unsigned short*)Bs[buf])[(2 * np + 1) * 32 + kg + j] = f2bf(v1);
      }
    }
  };

  v8f acc[2][2];
  #pragma unroll
  for (int i = 0; i < 2; ++i)
    #pragma unroll
    for (int j = 0; j < 2; ++j)
      acc[i][j] = (v8f){0.f,0.f,0.f,0.f,0.f,0.f,0.f,0.f};

  const int nk = (K + GBK - 1) / GBK;
  stageA(0, 0);
  stageB(0, 0);

  for (int kc = 0; kc < nk; ++kc) {
    __syncthreads();
    const int cur = kc & 1;
    if (kc + 1 < nk) {             // prefetch next chunk into other buffer
      stageA(cur ^ 1, (kc + 1) * GBK);
      stageB(cur ^ 1, (kc + 1) * GBK);
    }
    if (kc + 2 < nk && (m0 + GBM <= M) && (n0 + GBN <= N)) {
      // pull the chunk-after-next toward the WGP (global_prefetch_b8)
      const int kp = (kc + 2) * GBK;
      __builtin_prefetch(A + (size_t)(m0 + row_a) * lda + kp + ka, 0, 3);
      __builtin_prefetch(W + (size_t)(n0 + 2 * np) * ldw + kp + kg, 0, 3);
    }
    const unsigned int* Ab = As[cur];
    const unsigned int* Bb = Bs[cur];
    FragU af[2], bf_[2];
    #pragma unroll
    for (int i = 0; i < 2; ++i) {
      const int base = (wm * 32 + i * 16 + mrow) * 16 + hi * 4;
      af[i].h[0] = *(const u16x8*)&Ab[base];
      af[i].h[1] = *(const u16x8*)&Ab[base + 8];
    }
    #pragma unroll
    for (int j = 0; j < 2; ++j) {
      const int base = (wn * 32 + j * 16 + mrow) * 16 + hi * 4;
      bf_[j].h[0] = *(const u16x8*)&Bb[base];
      bf_[j].h[1] = *(const u16x8*)&Bb[base + 8];
    }
    #pragma unroll
    for (int i = 0; i < 2; ++i)
      #pragma unroll
      for (int j = 0; j < 2; ++j)
        acc[i][j] = wmma_bf16(af[i], bf_[j], acc[i][j]);
  }

  // epilogue: C layout — N = lane&15, M = r + 8*(lane>>4)
  const int cn = lane & 15, cm = (lane >> 4) * 8;
  if ((m0 + GBM <= M) && (n0 + GBN <= N)) {     // interior: unconditional
    #pragma unroll
    for (int i = 0; i < 2; ++i) {
      #pragma unroll
      for (int j = 0; j < 2; ++j) {
        const int nabs = n0 + wn * 32 + j * 16 + cn;
        const float bv = bias ? bias[nabs] : 0.f;
        float* cp = C + (size_t)(m0 + wm * 32 + i * 16 + cm) * ldc + nabs;
        #pragma unroll
        for (int r = 0; r < 8; ++r) {
          float v = acc[i][j][r] + bv;
          if (relu) v = fmaxf(v, 0.f);
          cp[(size_t)r * ldc] = v;
        }
      }
    }
  } else {
    #pragma unroll
    for (int i = 0; i < 2; ++i) {
      #pragma unroll
      for (int j = 0; j < 2; ++j) {
        const int nabs = n0 + wn * 32 + j * 16 + cn;
        if (nabs < N) {
          const float bv = bias ? bias[nabs] : 0.f;
          #pragma unroll
          for (int r = 0; r < 8; ++r) {
            const int mabs = m0 + wm * 32 + i * 16 + cm + r;
            if (mabs < M) {
              float v = acc[i][j][r] + bv;
              if (relu) v = fmaxf(v, 0.f);
              C[(size_t)mabs * ldc + nabs] = v;
            }
          }
        }
      }
    }
  }
}

// ---------------------------------------------------------------------------
// Prep kernels
// ---------------------------------------------------------------------------
__global__ void k_addbias(const float* __restrict__ a, const float* __restrict__ b,
                          float* __restrict__ o, int n) {
  const int i = blockIdx.x * 256 + threadIdx.x;
  if (i < n) o[i] = a[i] + b[i];
}

// Whh[1200,300] -> Whh bf16 [1200,320(pad)], pad cols zeroed
__global__ void k_whhT(const float* __restrict__ whh, unsigned short* __restrict__ wt) {
  const int i = blockIdx.x * 256 + threadIdx.x;
  if (i >= 320 * 1200) return;
  const int n = i / 320, k = i % 320;
  wt[i] = (k < 300) ? f2bf(whh[n * 300 + k]) : (unsigned short)0;
}

// ---------------------------------------------------------------------------
// Persistent bidirectional LSTM recurrence: 2 blocks (dir), 1024 threads.
// gates[8,1200] = h_pad[16x320] * WhhT  (WMMA, bf16)  + xg[t], then pointwise.
// gates has a 9th dummy row so the C-tile writeback is branch-free.
// ---------------------------------------------------------------------------
__global__ __launch_bounds__(1024) void k_lstm(
    const float* __restrict__ xg_f, const float* __restrict__ xg_b,
    const unsigned short* __restrict__ wt_f, const unsigned short* __restrict__ wt_b,
    float* __restrict__ hout) {
  const int dir = blockIdx.x;
  const float* xg = dir ? xg_b : xg_f;
  const unsigned short* WT = dir ? wt_b : wt_f;

  __shared__ unsigned short hA[16 * 320];  // h as bf16 A-tile, padded M/K
  __shared__ float gates[9 * 1200];        // rows 0..7 real, row 8 = dummy sink
  __shared__ float cbuf[8 * 300];

  const int tid = threadIdx.x, wave = tid >> 5, lane = tid & 31;
  for (int i = tid; i < 16 * 320; i += 1024) hA[i] = 0;
  for (int i = tid; i < 8 * 300; i += 1024) cbuf[i] = 0.f;
  __syncthreads();

  const int mrow = lane & 15, hi = lane >> 4;
  const int cn = lane & 15, cm = (lane >> 4) * 8;

  for (int t = 0; t < 128; ++t) {
    const int l = dir ? (127 - t) : t;

    // recurrent GEMM: 75 N-tiles of 16, K = 320 (10 chunks of 32)
    for (int tile = wave; tile < 75; tile += 32) {   // wave-uniform
      const int n0t = tile * 16;
      v8f acc = (v8f){0.f,0.f,0.f,0.f,0.f,0.f,0.f,0.f};
      #pragma unroll
      for (int kc = 0; kc < 10; ++kc) {
        const int kb = kc * 32;
        FragU af, bfr;
        af.h[0] = *(const u16x8*)&hA[mrow * 320 + kb + hi * 8];
        af.h[1] = *(const u16x8*)&hA[mrow * 320 + kb + 16 + hi * 8];
        const u16x8* q = (const u16x8*)(WT + (size_t)(n0t + mrow) * 320 + kb + hi * 8);
        bfr.h[0] = q[0]; bfr.h[1] = q[2];
        acc = wmma_bf16(af, bfr, acc);
      }
      #pragma unroll
      for (int r = 0; r < 8; ++r) {
        const int m = cm + r;                   // 0..7 (lanes 0-15), 8..15 (16-31)
        const int mg = m < 8 ? m : 8;           // dummy row sink, branch-free
        const float xv = xg[((size_t)(m & 7) * 128 + l) * 1200 + n0t + cn];
        gates[mg * 1200 + n0t + cn] = acc[r] + xv;
      }
    }
    __syncthreads();

    // pointwise LSTM cell (gate order i,f,g,o)
    for (int idx = tid; idx < 2400; idx += 1024) {
      const int m = idx / 300, n = idx % 300;
      const float gi = gates[m * 1200 + n];
      const float gf = gates[m * 1200 + 300 + n];
      const float gg = gates[m * 1200 + 600 + n];
      const float go = gates[m * 1200 + 900 + n];
      const float si = 1.f / (1.f + __expf(-gi));
      const float sf = 1.f / (1.f + __expf(-gf));
      const float so = 1.f / (1.f + __expf(-go));
      const float c  = sf * cbuf[idx] + si * tanhf(gg);
      const float h  = so * tanhf(c);
      cbuf[idx] = c;
      hA[m * 320 + n] = f2bf(h);
      hout[((size_t)m * 128 + l) * 600 + dir * 300 + n] = h;
    }
    __syncthreads();
  }
}

// ---------------------------------------------------------------------------
// Span assembly: spans[b,s,:] = [h[start]; h[end]; width_emb[w]]
// ---------------------------------------------------------------------------
__global__ __launch_bounds__(256) void k_spans(
    const float* __restrict__ hout, const float* __restrict__ wemb,
    float* __restrict__ spans) {
  const int bs = blockIdx.x;             // 0..5039
  const int b = bs / 630, s = bs % 630;
  int st, en; span_se(s, &st, &en);
  const int w = en - st + 1;
  float* dst = spans + (size_t)bs * 1220;
  const float* hs = hout + ((size_t)b * 128 + st) * 600;
  const float* he = hout + ((size_t)b * 128 + en) * 600;
  for (int t = threadIdx.x; t < 1220; t += 256) {
    float v;
    if (t < 600)       v = hs[t];
    else if (t < 1200) v = he[t - 600];
    else               v = wemb[w * 20 + (t - 1200)];
    dst[t] = v;
  }
}

// ---------------------------------------------------------------------------
// Heads: small-N GEMM + softmax (N=3 spans, N=4 pairs)
// ---------------------------------------------------------------------------
__global__ __launch_bounds__(256) void k_span_head(
    const float* __restrict__ h2, const float* __restrict__ w3,
    const float* __restrict__ b3, float* __restrict__ sp_ws,
    float* __restrict__ sp_out) {
  __shared__ float w[3 * 150];
  __shared__ float bb[3];
  const int tid = threadIdx.x;
  for (int i = tid; i < 450; i += 256) w[i] = w3[i];
  if (tid < 3) bb[tid] = b3[tid];
  __syncthreads();
  const int r = blockIdx.x * 256 + tid;
  if (r >= 5040) return;
  float z0 = bb[0], z1 = bb[1], z2 = bb[2];
  const float* hr = h2 + (size_t)r * 150;
  for (int k = 0; k < 150; ++k) {
    const float v = hr[k];
    z0 += v * w[k]; z1 += v * w[150 + k]; z2 += v * w[300 + k];
  }
  const float mx = fmaxf(z0, fmaxf(z1, z2));
  const float e0 = __expf(z0 - mx), e1 = __expf(z1 - mx), e2 = __expf(z2 - mx);
  const float inv = 1.f / (e0 + e1 + e2);
  sp_ws[r * 3 + 0] = e0 * inv; sp_ws[r * 3 + 1] = e1 * inv; sp_ws[r * 3 + 2] = e2 * inv;
  sp_out[r * 3 + 0] = e0 * inv; sp_out[r * 3 + 1] = e1 * inv; sp_out[r * 3 + 2] = e2 * inv;
}

__global__ __launch_bounds__(256) void k_pair_head(
    const float* __restrict__ c2, const float* __restrict__ w3,
    const float* __restrict__ b3, float* __restrict__ outp) {
  __shared__ float w[4 * 150];
  __shared__ float bb[4];
  const int tid = threadIdx.x;
  for (int i = tid; i < 600; i += 256) w[i] = w3[i];
  if (tid < 4) bb[tid] = b3[tid];
  __syncthreads();
  const int r = blockIdx.x * 256 + tid;   // 0..32767
  float z[4] = {bb[0], bb[1], bb[2], bb[3]};
  const float* hr = c2 + (size_t)r * 150;
  for (int k = 0; k < 150; ++k) {
    const float v = hr[k];
    #pragma unroll
    for (int c = 0; c < 4; ++c) z[c] += v * w[c * 150 + k];
  }
  float mx = z[0];
  #pragma unroll
  for (int c = 1; c < 4; ++c) mx = fmaxf(mx, z[c]);
  float e[4], sum = 0.f;
  #pragma unroll
  for (int c = 0; c < 4; ++c) { e[c] = __expf(z[c] - mx); sum += e[c]; }
  const float inv = 1.f / sum;
  #pragma unroll
  for (int c = 0; c < 4; ++c) outp[(size_t)r * 4 + c] = e[c] * inv;
}

// ---------------------------------------------------------------------------
// Top-64 per (batch, class) by iterative argmax (stable: lower index wins ties)
// ---------------------------------------------------------------------------
__global__ __launch_bounds__(256) void k_topk(
    const float* __restrict__ sprob, int* __restrict__ tout, int* __restrict__ oout) {
  const int b = blockIdx.x;
  const int cls = blockIdx.y + 1;        // 1 = aspect, 2 = opinion
  __shared__ float sv[630];
  __shared__ float rv[256];
  __shared__ int   ri[256];
  const int tid = threadIdx.x;
  for (int s = tid; s < 630; s += 256)
    sv[s] = sprob[((size_t)b * 630 + s) * 3 + cls];
  __syncthreads();
  int* outp = (cls == 1) ? tout : oout;
  for (int it = 0; it < 64; ++it) {
    float bv = -1e30f; int bi = 0x7fffffff;
    for (int s = tid; s < 630; s += 256) {
      const float v = sv[s];
      if (v > bv || (v == bv && s < bi)) { bv = v; bi = s; }
    }
    rv[tid] = bv; ri[tid] = bi;
    __syncthreads();
    for (int off = 128; off > 0; off >>= 1) {
      if (tid < off) {
        const float v2 = rv[tid + off]; const int i2 = ri[tid + off];
        if (v2 > rv[tid] || (v2 == rv[tid] && i2 < ri[tid])) { rv[tid] = v2; ri[tid] = i2; }
      }
      __syncthreads();
    }
    if (tid == 0) { outp[b * 64 + it] = ri[0]; sv[ri[0]] = -1e30f; }
    __syncthreads();
  }
}

// gather pruned span rows into packed [512,1220]
__global__ __launch_bounds__(256) void k_gather(
    const float* __restrict__ spans, const int* __restrict__ idx,
    float* __restrict__ dst) {
  const int row = blockIdx.x;            // 0..511
  const int b = row >> 6;
  const int s = idx[row];
  const float* src = spans + ((size_t)b * 630 + s) * 1220;
  float* d = dst + (size_t)row * 1220;
  for (int t = threadIdx.x; t < 1220; t += 256) d[t] = src[t];
}

// pair layer-1 assembly: relu(projT[i] + projO[j] + projD[dist])  (pb1 in projD)
__global__ __launch_bounds__(160) void k_pair_l1(
    const float* __restrict__ pT, const float* __restrict__ pO,
    const float* __restrict__ pD, const int* __restrict__ tix,
    const int* __restrict__ oix, float* __restrict__ cand1) {
  const int p = blockIdx.x;              // 0..32767
  const int b = p >> 12, i = (p >> 6) & 63, j = p & 63;
  const int ts = tix[b * 64 + i], os = oix[b * 64 + j];
  int a0, b0, c0, d0;
  span_se(ts, &a0, &b0); span_se(os, &c0, &d0);
  int d1 = b0 - c0; if (d1 < 0) d1 = -d1;
  int d2 = a0 - d0; if (d2 < 0) d2 = -d2;
  const int dist = d1 < d2 ? d1 : d2;
  const int f = threadIdx.x;
  if (f < 150) {
    const float v = pT[(size_t)(b * 64 + i) * 150 + f] +
                    pO[(size_t)(b * 64 + j) * 150 + f] +
                    pD[(size_t)dist * 150 + f];
    cand1[(size_t)p * 150 + f] = fmaxf(v, 0.f);
  }
}

__global__ void k_write_idx(const int* __restrict__ t, const int* __restrict__ o,
                            float* __restrict__ outp) {
  const int i = blockIdx.x * 256 + threadIdx.x;
  if (i < 512)        outp[i] = (float)t[i];
  else if (i < 1024)  outp[i] = (float)o[i - 512];
}

// ---------------------------------------------------------------------------
// Launch
// ---------------------------------------------------------------------------
extern "C" void kernel_launch(void* const* d_in, const int* in_sizes, int n_in,
                              void* d_out, int out_size, void* d_ws, size_t ws_size,
                              hipStream_t stream) {
  (void)in_sizes; (void)n_in; (void)out_size; (void)ws_size;
  const float* x        = (const float*)d_in[0];
  const float* Wih_f    = (const float*)d_in[1];
  const float* Whh_f    = (const float*)d_in[2];
  const float* bih_f    = (const float*)d_in[3];
  const float* bhh_f    = (const float*)d_in[4];
  const float* Wih_b    = (const float*)d_in[5];
  const float* Whh_b    = (const float*)d_in[6];
  const float* bih_b    = (const float*)d_in[7];
  const float* bhh_b    = (const float*)d_in[8];
  const float* wemb     = (const float*)d_in[9];
  const float* demb     = (const float*)d_in[10];
  const float* sw1      = (const float*)d_in[11];
  const float* sb1      = (const float*)d_in[12];
  const float* sw2      = (const float*)d_in[13];
  const float* sb2      = (const float*)d_in[14];
  const float* sw3      = (const float*)d_in[15];
  const float* sb3      = (const float*)d_in[16];
  const float* pw1      = (const float*)d_in[17];
  const float* pb1      = (const float*)d_in[18];
  const float* pw2      = (const float*)d_in[19];
  const float* pb2      = (const float*)d_in[20];
  const float* pw3      = (const float*)d_in[21];
  const float* pb3      = (const float*)d_in[22];

  char* wsb = (char*)d_ws;
  size_t off = 0;
  auto alloc = [&](size_t bytes) -> void* {
    off = (off + 255) & ~(size_t)255;
    void* p = wsb + off;
    off += bytes;
    return p;
  };
  float* xg_f  = (float*)alloc((size_t)1024 * 1200 * 4);
  float* xg_b  = (float*)alloc((size_t)1024 * 1200 * 4);
  float* bias_f = (float*)alloc(1200 * 4);
  float* bias_b = (float*)alloc(1200 * 4);
  unsigned short* wt_f = (unsigned short*)alloc((size_t)320 * 1200 * 2);
  unsigned short* wt_b = (unsigned short*)alloc((size_t)320 * 1200 * 2);
  float* hout  = (float*)alloc((size_t)8 * 128 * 600 * 4);
  float* spans = (float*)alloc((size_t)5040 * 1220 * 4);
  float* h1    = (float*)alloc((size_t)5040 * 150 * 4);
  float* h2    = (float*)alloc((size_t)5040 * 150 * 4);
  float* sprob = (float*)alloc((size_t)5040 * 3 * 4);
  int*   tix   = (int*)alloc(512 * 4);
  int*   oix   = (int*)alloc(512 * 4);
  float* tsp   = (float*)alloc((size_t)512 * 1220 * 4);
  float* osp   = (float*)alloc((size_t)512 * 1220 * 4);
  float* pT    = (float*)alloc((size_t)512 * 150 * 4);
  float* pO    = (float*)alloc((size_t)512 * 150 * 4);
  float* pD    = (float*)alloc((size_t)512 * 150 * 4);
  float* c1    = (float*)alloc((size_t)32768 * 150 * 4);
  float* c2    = (float*)alloc((size_t)32768 * 150 * 4);

  auto ggrid = [](int M, int N) { return dim3((N + GBN - 1) / GBN, (M + GBM - 1) / GBM); };

  // prep
  k_addbias<<<(1200 + 255) / 256, 256, 0, stream>>>(bih_f, bhh_f, bias_f, 1200);
  k_addbias<<<(1200 + 255) / 256, 256, 0, stream>>>(bih_b, bhh_b, bias_b, 1200);
  k_whhT<<<(320 * 1200 + 255) / 256, 256, 0, stream>>>(Whh_f, wt_f);
  k_whhT<<<(320 * 1200 + 255) / 256, 256, 0, stream>>>(Whh_b, wt_b);

  // LSTM input projections (bf16 WMMA): [1024,768] x [768,1200]
  k_gemm<<<ggrid(1024, 1200), 256, 0, stream>>>(x, Wih_f, bias_f, xg_f,
      1024, 1200, 768, 768, 768, 1200, 0);
  k_gemm<<<ggrid(1024, 1200), 256, 0, stream>>>(x, Wih_b, bias_b, xg_b,
      1024, 1200, 768, 768, 768, 1200, 0);

  // recurrence (persistent, one block per direction)
  k_lstm<<<2, 1024, 0, stream>>>(xg_f, xg_b, wt_f, wt_b, hout);

  // spans + span FFNN
  k_spans<<<5040, 256, 0, stream>>>(hout, wemb, spans);
  k_gemm<<<ggrid(5040, 150), 256, 0, stream>>>(spans, sw1, sb1, h1,
      5040, 150, 1220, 1220, 1220, 150, 1);
  k_gemm<<<ggrid(5040, 150), 256, 0, stream>>>(h1, sw2, sb2, h2,
      5040, 150, 150, 150, 150, 150, 1);
  k_span_head<<<(5040 + 255) / 256, 256, 0, stream>>>(h2, sw3, sb3, sprob, (float*)d_out);

  // prune + pair layer-1 factorization
  k_topk<<<dim3(8, 2), 256, 0, stream>>>(sprob, tix, oix);
  k_gather<<<512, 256, 0, stream>>>(spans, tix, tsp);
  k_gather<<<512, 256, 0, stream>>>(spans, oix, osp);
  k_gemm<<<ggrid(512, 150), 256, 0, stream>>>(tsp, pw1, nullptr, pT,
      512, 150, 1220, 1220, 2568, 150, 0);
  k_gemm<<<ggrid(512, 150), 256, 0, stream>>>(osp, pw1 + 1220, nullptr, pO,
      512, 150, 1220, 1220, 2568, 150, 0);
  k_gemm<<<ggrid(512, 150), 256, 0, stream>>>(demb, pw1 + 2440, pb1, pD,
      512, 150, 128, 128, 2568, 150, 0);
  k_pair_l1<<<32768, 160, 0, stream>>>(pT, pO, pD, tix, oix, c1);

  // pair FFNN layers 2..3
  k_gemm<<<ggrid(32768, 150), 256, 0, stream>>>(c1, pw2, pb2, c2,
      32768, 150, 150, 150, 150, 150, 1);
  k_pair_head<<<32768 / 256, 256, 0, stream>>>(c2, pw3, pb3, (float*)d_out + 15120);

  // indices
  k_write_idx<<<4, 256, 0, stream>>>(tix, oix, (float*)d_out + 15120 + 131072);
}
